// GATLayer_72267119722746
// MI455X (gfx1250) — matrix operand order repaired and checked by reading, not compiled, not run on hardware
//
#include <hip/hip_runtime.h>
#include <hip/hip_bf16.h>

// GAT layer: B=128, M=512, in_dim=128, out_dim=64, fp32 in/out.
// Kernel 1: Wh = h @ W^T (f16 WMMA, f32 acc), emit Wh^T (f16) + s_src/s_dst (f32).
// Kernel 2: stage Wh^T[b] in LDS (async global->LDS, ASYNCcnt), fused
//           leakyrelu+softmax+GEMM2 (attn @ Wh) + ELU, WMMA f16.

#define B_BATCH 128
#define M_NODES 512
#define IN_DIM  128
#define OUT_DIM 64
#define ALPHA_L 0.2f
#define WPAD 144   // kernel1 LDS row stride (halfs) for W
#define WHPAD 520  // kernel2 LDS row stride (halfs) for Wh^T: 260 dwords = 4 mod 64 banks

#define AS(n) __attribute__((address_space(n)))

#if defined(__has_builtin)
#if __has_builtin(__builtin_amdgcn_global_load_async_to_lds_b128) && \
    __has_builtin(__builtin_amdgcn_s_wait_asynccnt)
#define USE_ASYNC_LDS 1
#endif
#endif

typedef __attribute__((ext_vector_type(16))) _Float16 v16h;
typedef __attribute__((ext_vector_type(8)))  _Float16 v8h;
typedef __attribute__((ext_vector_type(8)))  float    v8f;
typedef __attribute__((ext_vector_type(4)))  float    v4f;
typedef __attribute__((ext_vector_type(4)))  int      v4i;

__device__ __forceinline__ float lrelu(float x) { return x > 0.f ? x : ALPHA_L * x; }

// ---------------------------------------------------------------------------
// Kernel 1: projection. grid = (B*M)/64 workgroups, 128 threads (4 waves).
// Each wave: 16 rows x 64 cols, K=128 in 4 steps of 32 via v_wmma_f32_16x16x32_f16.
// ---------------------------------------------------------------------------
__global__ __launch_bounds__(128) void gat_proj(
    const float* __restrict__ h, const float* __restrict__ W,
    const float* __restrict__ a,
    _Float16* __restrict__ whT, float* __restrict__ ssrc, float* __restrict__ sdst)
{
  __shared__ alignas(32) _Float16 wlds[OUT_DIM * WPAD];

  const int tid  = threadIdx.x;
  const int lane = tid & 31;
  const int wave = tid >> 5;

  // Preload W (64x128 f32) into LDS as f16 with padded stride.
  for (int idx = tid; idx < OUT_DIM * IN_DIM; idx += 128) {
    int d = idx >> 7, i = idx & 127;
    wlds[d * WPAD + i] = (_Float16)W[idx];
  }
  __syncthreads();

  const int rowt = blockIdx.x * 64 + wave * 16; // tile base in flattened (b*M+m)
  const int half = lane >> 4;                   // ISA A/B layout: lane half
  const int lid  = lane & 15;
  const int rowA = rowt + lid;                  // A-fragment row for this lane
  const int kbA  = half * 16;                   // A: K base 0/16 per half
  const int kbB  = half * 16;                   // B: K base 0/16 per half

  v8f c[4] = {};
  for (int kk = 0; kk < IN_DIM; kk += 32) {
    // A fragment: h[rowA, kk+kbA+0..15], f32->f16
    const float* hp = h + (size_t)rowA * IN_DIM + kk + kbA;
    v4f h0 = *(const v4f*)(hp);
    v4f h1 = *(const v4f*)(hp + 4);
    v4f h2 = *(const v4f*)(hp + 8);
    v4f h3 = *(const v4f*)(hp + 12);
    v16h aF;
    aF[0]=(_Float16)h0[0]; aF[1]=(_Float16)h0[1]; aF[2]=(_Float16)h0[2]; aF[3]=(_Float16)h0[3];
    aF[4]=(_Float16)h1[0]; aF[5]=(_Float16)h1[1]; aF[6]=(_Float16)h1[2]; aF[7]=(_Float16)h1[3];
    aF[8]=(_Float16)h2[0]; aF[9]=(_Float16)h2[1]; aF[10]=(_Float16)h2[2]; aF[11]=(_Float16)h2[3];
    aF[12]=(_Float16)h3[0]; aF[13]=(_Float16)h3[1]; aF[14]=(_Float16)h3[2]; aF[15]=(_Float16)h3[3];
#pragma unroll
    for (int t = 0; t < 4; ++t) {
      const v16h bF = *(const v16h*)(&wlds[(t * 16 + lid) * WPAD + kk + kbB]);
      c[t] = __builtin_amdgcn_wmma_f32_16x16x32_f16(
          false, aF, false, bF, (short)0, c[t], false, false);
    }
  }

  // Epilogue: store Wh^T (f16, d-major) and reduce s_src/s_dst.
  const int b = rowt >> 9; // all 16 rows share one batch (512 % 16 == 0)
  float asv[4], adv[4];
#pragma unroll
  for (int t = 0; t < 4; ++t) {
    asv[t] = a[t * 16 + lid];
    adv[t] = a[OUT_DIM + t * 16 + lid];
  }
  float ps[8] = {}, pd[8] = {};
#pragma unroll
  for (int t = 0; t < 4; ++t) {
    const int d = t * 16 + lid;
    const size_t dbase = ((size_t)(b * OUT_DIM + d) << 9);
#pragma unroll
    for (int j = 0; j < 8; ++j) {
      float v = c[t][j];
      const int m = (rowt + half * 8 + j) & (M_NODES - 1);
      whT[dbase + m] = (_Float16)v;
      ps[j] += v * asv[t];
      pd[j] += v * adv[t];
    }
  }
  // Butterfly over the 16 lanes of each half (columns d of one tile row group).
#pragma unroll
  for (int msk = 8; msk >= 1; msk >>= 1) {
#pragma unroll
    for (int j = 0; j < 8; ++j) {
      ps[j] += __shfl_xor(ps[j], msk, 32);
      pd[j] += __shfl_xor(pd[j], msk, 32);
    }
  }
  if (lid == 0) {
#pragma unroll
    for (int j = 0; j < 8; ++j) {
      const int r = rowt + half * 8 + j;
      ssrc[r] = ps[j];
      sdst[r] = pd[j];
    }
  }
}

// ---------------------------------------------------------------------------
// Kernel 2: fused softmax(LeakyReLU(s_src+s_dst)) @ Wh, then ELU.
// grid = B * (M/64), 128 threads (4 waves); wave does 16 rows x 64 cols, K=512.
// Wh^T[b] (64KB) staged in LDS once per WG via async global->LDS loads.
// Unnormalized exp goes into WMMA; row-sum accumulated in f32 alongside.
// ---------------------------------------------------------------------------
__global__ __launch_bounds__(128) void gat_attn(
    const _Float16* __restrict__ whT, const float* __restrict__ ssrc,
    const float* __restrict__ sdst, float* __restrict__ out)
{
  __shared__ alignas(16) _Float16 whl[OUT_DIM * WHPAD]; // 66,560 B, conflict-free stride
  __shared__ float sd[M_NODES];

  const int tid  = threadIdx.x;
  const int lane = tid & 31;
  const int wave = tid >> 5;
  const int b    = blockIdx.x >> 3;
  const int mblk = (blockIdx.x & 7) * 64;

  const _Float16* wb = whT + ((size_t)b * OUT_DIM << 9);

  // Stage Wh^T[b]: 64 rows x 512 halfs -> LDS stride WHPAD, 16B chunks.
  for (int cchunk = tid; cchunk < (OUT_DIM * M_NODES) / 8; cchunk += 128) {
    const int d  = cchunk >> 6;          // 64 chunks per row
    const int ko = (cchunk & 63) * 8;
#if defined(USE_ASYNC_LDS)
    __builtin_amdgcn_global_load_async_to_lds_b128(
        (AS(1) v4i*)(AS(1) void*)(wb + (d << 9) + ko),
        (AS(3) v4i*)(AS(3) void*)(&whl[d * WHPAD + ko]), 0, 0);
#else
    *(v8h*)(&whl[d * WHPAD + ko]) = *(const v8h*)(wb + (d << 9) + ko);
#endif
  }
  for (int i = tid; i < M_NODES; i += 128) sd[i] = sdst[b * M_NODES + i];
#if defined(USE_ASYNC_LDS)
  __builtin_amdgcn_s_wait_asynccnt(0);
#endif
  __syncthreads();

  // max_n s_dst (monotone LeakyReLU => safe softmax shift from this alone)
  float mx = -3.0e38f;
  for (int i = lane; i < M_NODES; i += 32) mx = fmaxf(mx, sd[i]);
#pragma unroll
  for (int msk = 16; msk >= 1; msk >>= 1) mx = fmaxf(mx, __shfl_xor(mx, msk, 32));

  const int half = lane >> 4;
  const int lid  = lane & 15;
  const int kbA  = half * 16;
  const int kbB  = half * 16;
  const int row  = mblk + wave * 16 + lid;
  const float ss = ssrc[b * M_NODES + row];
  const float zmax = lrelu(ss + mx);

  v8f c[4] = {};
  float rsum = 0.f;

  for (int nn = 0; nn < M_NODES; nn += 32) {
    v16h aF;
#pragma unroll
    for (int j = 0; j < 8; ++j) {
      float p = __expf(lrelu(ss + sd[nn + kbA + j]) - zmax);
      rsum += p;
      aF[j] = (_Float16)p;
    }
#pragma unroll
    for (int j = 0; j < 8; ++j) {
      float p = __expf(lrelu(ss + sd[nn + kbA + 8 + j]) - zmax);
      rsum += p;
      aF[8 + j] = (_Float16)p;
    }
#pragma unroll
    for (int t = 0; t < 4; ++t) {
      const _Float16* wrow = &whl[(t * 16 + lid) * WHPAD + nn + kbB];
      v8h b0 = *(const v8h*)(wrow);
      v8h b1 = *(const v8h*)(wrow + 8);
      v16h bF = __builtin_shufflevector(b0, b1, 0,1,2,3,4,5,6,7,8,9,10,11,12,13,14,15);
      c[t] = __builtin_amdgcn_wmma_f32_16x16x32_f16(
          false, aF, false, bF, (short)0, c[t], false, false);
    }
  }
  // Combine the two K-halves of each row (lanes l and l+16 share row l&15).
  rsum += __shfl_xor(rsum, 16, 32);

  // Normalize + ELU + store. C layout: vgpr j -> row j (lanes 0-15) / j+8 (16-31).
#pragma unroll
  for (int j = 0; j < 8; ++j) {
    const int rloc = half * 8 + j;                   // row-in-tile for c[*][j]
    const float inv = 1.0f / __shfl(rsum, rloc, 32); // rowsum lives at lane == row
    const int r = mblk + wave * 16 + rloc;
    const size_t obase = ((size_t)(b * M_NODES + r)) << 6;
#pragma unroll
    for (int t = 0; t < 4; ++t) {
      float v = c[t][j] * inv;
      v = v > 0.f ? v : (__expf(v) - 1.0f);          // ELU (alpha=1)
      out[obase + t * 16 + lid] = v;
    }
  }
}

// ---------------------------------------------------------------------------
extern "C" void kernel_launch(void* const* d_in, const int* in_sizes, int n_in,
                              void* d_out, int out_size, void* d_ws, size_t ws_size,
                              hipStream_t stream) {
  const float* h = (const float*)d_in[0];
  const float* W = (const float*)d_in[1];
  const float* a = (const float*)d_in[2];
  float* out = (float*)d_out;

  // Workspace: Wh^T f16 (8 MB) | s_src f32 (256 KB) | s_dst f32 (256 KB)
  _Float16* whT = (_Float16*)d_ws;
  const size_t whT_bytes = (size_t)B_BATCH * OUT_DIM * M_NODES * sizeof(_Float16);
  float* ssrc = (float*)((char*)d_ws + whT_bytes);
  float* sdst = ssrc + (size_t)B_BATCH * M_NODES;

  const int wg1 = (B_BATCH * M_NODES) / 64;   // 1024
  const int wg2 = B_BATCH * (M_NODES / 64);   // 1024
  gat_proj<<<wg1, 128, 0, stream>>>(h, W, a, whT, ssrc, sdst);
  gat_attn<<<wg2, 128, 0, stream>>>(whT, ssrc, sdst, out);
}
